// SelfAttentionND_58239756534131
// MI455X (gfx1250) — matrix operand order repaired and checked by reading, not compiled, not run on hardware
//
#include <hip/hip_runtime.h>
#include <hip/hip_bf16.h>

#define BATCH 2
#define SEQ   2048
#define DIM   1024
#define NH    16
#define HD    64
#define MTOT  (BATCH * SEQ)   // 4096
#define NHC   (NH * HD)       // 1024

typedef __attribute__((ext_vector_type(16))) __bf16 v16bf;
typedef __attribute__((ext_vector_type(8)))  float  v8f;

__device__ __forceinline__ v8f wmma_bf16(v16bf a, v16bf b, v8f c) {
  return __builtin_amdgcn_wmma_f32_16x16x32_bf16(
      false, a, false, b, (short)0, c, false, false);
}

// ---- A fragment: 16x32 bf16, interleaved-K layout (ISA 7.12.2, 16-bit A) ----
__device__ __forceinline__ v16bf load_a_frag(const __bf16* base, long row0, long k0, long ld) {
  int lane = threadIdx.x & 31;
  long r    = lane & 15;
  long half = lane >> 4;
  const __bf16* p = base + (row0 + r) * ld + k0 + half * 8;
  union { v16bf v; uint4 u[2]; } f;
  f.u[0] = *reinterpret_cast<const uint4*>(p);
  f.u[1] = *reinterpret_cast<const uint4*>(p + 16);
  return f.v;
}

// ---- B fragment from B^T stored row-major [N, K] (interleaved-K layout) ----
__device__ __forceinline__ v16bf load_b_frag(const __bf16* base, long n0, long k0, long ld) {
  int lane = threadIdx.x & 31;
  long r    = lane & 15;
  long half = lane >> 4;
  const __bf16* p = base + (n0 + r) * ld + k0 + half * 8;
  union { v16bf v; uint4 u[2]; } f;
  f.u[0] = *reinterpret_cast<const uint4*>(p);
  f.u[1] = *reinterpret_cast<const uint4*>(p + 16);
  return f.v;
}

// ---- global -> LDS 16-byte copy: async if toolchain exposes the builtins ----
#if __has_builtin(__builtin_amdgcn_global_load_async_to_lds_b128) && \
    __has_builtin(__builtin_amdgcn_s_wait_asynccnt)
#define USE_ASYNC_LDS 1
typedef int v4i __attribute__((vector_size(16)));
typedef __attribute__((address_space(1))) v4i* gv4i_p;   // global (AS1)
typedef __attribute__((address_space(3))) v4i* lv4i_p;   // LDS (AS3)
__device__ __forceinline__ void cp16_g2l(const __bf16* g, __bf16* l) {
  // generic LDS address: low 32 bits are the LDS offset (ISA 10.2 aperture rule)
  __builtin_amdgcn_global_load_async_to_lds_b128(
      (gv4i_p)(unsigned long long)g,
      (lv4i_p)(unsigned int)(unsigned long long)l, 0, 0);
}
__device__ __forceinline__ void cp_wait_all() { __builtin_amdgcn_s_wait_asynccnt(0); }
#else
#define USE_ASYNC_LDS 0
__device__ __forceinline__ void cp16_g2l(const __bf16* g, __bf16* l) {
  *reinterpret_cast<uint4*>(l) = *reinterpret_cast<const uint4*>(g);
}
__device__ __forceinline__ void cp_wait_all() {}
#endif

// ---------------- conversion / transpose helpers ----------------
__global__ void cvt_f32_to_bf16(const float* __restrict__ in, __bf16* __restrict__ out, int n) {
  int i = blockIdx.x * blockDim.x + threadIdx.x;
  if (i < n) out[i] = (__bf16)in[i];
}

__global__ void transpose_cvt(const float* __restrict__ src, __bf16* __restrict__ dst) {
  __shared__ float tile[32][33];
  int bx = blockIdx.x * 32, by = blockIdx.y * 32;
  int tx = threadIdx.x, ty = threadIdx.y;
#pragma unroll
  for (int i = 0; i < 32; i += 8)
    tile[ty + i][tx] = src[(long)(by + ty + i) * DIM + bx + tx];
  __syncthreads();
#pragma unroll
  for (int i = 0; i < 32; i += 8)
    dst[(long)(bx + ty + i) * DIM + by + tx] = (__bf16)tile[tx][ty + i];
}

// ---------------- fused QKV projection (32x64 register tile / wave) ----------------
__global__ void qkv_proj(const __bf16* __restrict__ xb,
                         const __bf16* __restrict__ wqt,
                         const __bf16* __restrict__ wkt,
                         const __bf16* __restrict__ wvt,
                         __bf16* __restrict__ qb, __bf16* __restrict__ kb,
                         __bf16* __restrict__ vT) {
  int wave = threadIdx.x >> 5;
  int mt = wave >> 2, ntw = wave & 3;
  long row0 = (long)blockIdx.y * 64 + mt * 32;      // 2 waves x 32 rows
  long col0 = (long)blockIdx.x * 256 + ntw * 64;    // 4 waves x 64 cols
  int z = blockIdx.z;
  const __bf16* W = (z == 0) ? wqt : ((z == 1) ? wkt : wvt);

  v8f acc[2][4] = {};
  for (int k0 = 0; k0 < DIM; k0 += 32) {
    v16bf a0 = load_a_frag(xb, row0, k0, DIM);
    v16bf a1 = load_a_frag(xb, row0 + 16, k0, DIM);
    __builtin_prefetch(W + col0 * DIM + k0 + 256, 0, 1);
#pragma unroll
    for (int j = 0; j < 4; ++j) {
      v16bf bfr = load_b_frag(W, col0 + j * 16, k0, DIM);
      acc[0][j] = wmma_bf16(a0, bfr, acc[0][j]);
      acc[1][j] = wmma_bf16(a1, bfr, acc[1][j]);
    }
  }

  int lane = threadIdx.x & 31;
  int half = lane >> 4;
  int cidx = lane & 15;
#pragma unroll
  for (int i = 0; i < 2; ++i) {
#pragma unroll
    for (int j = 0; j < 4; ++j) {
#pragma unroll
      for (int r = 0; r < 8; ++r) {
        long m = row0 + i * 16 + r + half * 8;     // b*SEQ + s
        long n = col0 + j * 16 + cidx;             // h*HD + c
        long bidx = m >> 11, s = m & (SEQ - 1);
        long h = n >> 6, cc = n & (HD - 1);
        float val = acc[i][j][r];
        if (z == 0)
          qb[(((bidx * NH + h) * SEQ + s) * HD) + cc] = (__bf16)(val * 0.125f);
        else if (z == 1)
          kb[(((bidx * NH + h) * SEQ + s) * HD) + cc] = (__bf16)val;
        else
          vT[(((bidx * NH + h) * HD + cc) * SEQ) + s] = (__bf16)val;
      }
    }
  }
}

// ---------------- flash attention: LDS-shared K/V tiles, 64 keys / iter ----------------
__global__ void attention(const __bf16* __restrict__ qb,
                          const __bf16* __restrict__ kb,
                          const __bf16* __restrict__ vT,
                          __bf16* __restrict__ att) {
  __shared__ __align__(16) __bf16 Kt[64 * 64];        // [key][c]   8 KB
  __shared__ __align__(16) __bf16 Vt[64 * 64];        // [c][key]   8 KB
  __shared__ __align__(16) __bf16 pbuf[8][16 * 64];   // per-wave P 16 KB

  int tid = threadIdx.x;
  int bh = blockIdx.y;
  long b = bh >> 4, h = bh & 15;
  int wave = tid >> 5;
  long q0 = (long)blockIdx.x * 128 + wave * 16;

  const __bf16* Q = qb + (long)bh * SEQ * HD;
  const __bf16* K = kb + (long)bh * SEQ * HD;
  const __bf16* V = vT + (long)bh * HD * SEQ;
  __bf16* myp = pbuf[wave];

  int lane = tid & 31;
  int half = lane >> 4;
  int cidx = lane & 15;

  v16bf qf0 = load_a_frag(Q, q0, 0, HD);
  v16bf qf1 = load_a_frag(Q, q0, 32, HD);

  v8f O[4] = {};
  float mrow[8], lrow[8];
#pragma unroll
  for (int r = 0; r < 8; ++r) { mrow[r] = -__builtin_inff(); lrow[r] = 0.f; }

  for (int kblk = 0; kblk < SEQ; kblk += 64) {
    __syncthreads();  // previous tile fully consumed
    // cooperative staging: K tile is 8 KB contiguous; V^T tile is 64 rows x 128 B
#pragma unroll
    for (int i = 0; i < 2; ++i) {
      int ch = tid + i * 256;               // 512 x 16B chunks each
      cp16_g2l(K + (long)kblk * HD + ch * 8, Kt + ch * 8);
      int row = ch >> 3, part = ch & 7;
      cp16_g2l(V + (long)row * SEQ + kblk + part * 8, Vt + row * 64 + part * 8);
    }
    cp_wait_all();
    __syncthreads();

    // scores for 64 keys: 4 x 16x16 tiles
    v8f S[4] = {};
#pragma unroll
    for (int t = 0; t < 4; ++t) {
      v16bf kf0 = load_b_frag(Kt, t * 16, 0, HD);
      v16bf kf1 = load_b_frag(Kt, t * 16, 32, HD);
      S[t] = wmma_bf16(qf0, kf0, S[t]);
      S[t] = wmma_bf16(qf1, kf1, S[t]);
    }

    // online softmax across the 64 new keys (row = r + half*8, 16 lanes/row)
#pragma unroll
    for (int r = 0; r < 8; ++r) {
      float t0 = fmaxf(fmaxf(S[0][r], S[1][r]), fmaxf(S[2][r], S[3][r]));
#pragma unroll
      for (int mask = 8; mask >= 1; mask >>= 1)
        t0 = fmaxf(t0, __shfl_xor(t0, mask, 16));
      float mnew = fmaxf(mrow[r], t0);
      float corr = __expf(mrow[r] - mnew);
      mrow[r] = mnew;
      float ps = 0.f;
#pragma unroll
      for (int t = 0; t < 4; ++t) {
        float p = __expf(S[t][r] - mnew);
        S[t][r] = p;
        ps += p;
      }
#pragma unroll
      for (int mask = 8; mask >= 1; mask >>= 1)
        ps += __shfl_xor(ps, mask, 16);
      lrow[r] = lrow[r] * corr + ps;
      O[0][r] *= corr; O[1][r] *= corr; O[2][r] *= corr; O[3][r] *= corr;
    }

    // P (16x64, C-layout) -> LDS row-major -> interleaved A fragments
#pragma unroll
    for (int r = 0; r < 8; ++r) {
      int rowi = r + half * 8;
#pragma unroll
      for (int t = 0; t < 4; ++t)
        myp[rowi * 64 + t * 16 + cidx] = (__bf16)S[t][r];
    }
    v16bf pf0 = load_a_frag(myp, 0, 0, 64);
    v16bf pf1 = load_a_frag(myp, 0, 32, 64);

    // O += P(16x64) * V(64xHD) from LDS V^T tile
#pragma unroll
    for (int nt = 0; nt < 4; ++nt) {
      v16bf vf0 = load_b_frag(Vt, nt * 16, 0, 64);
      v16bf vf1 = load_b_frag(Vt, nt * 16, 32, 64);
      O[nt] = wmma_bf16(pf0, vf0, O[nt]);
      O[nt] = wmma_bf16(pf1, vf1, O[nt]);
    }
  }

  // normalize; write att[b][s][h][c]
#pragma unroll
  for (int r = 0; r < 8; ++r) {
    float inv = 1.0f / lrow[r];
    long s = q0 + r + half * 8;
#pragma unroll
    for (int nt = 0; nt < 4; ++nt) {
      long cc = nt * 16 + cidx;
      att[(((b * SEQ + s) * NH + h) * HD) + cc] = (__bf16)(O[nt][r] * inv);
    }
  }
}

// ---------------- output projection (32x64 register tile / wave) ----------------
__global__ void out_proj(const __bf16* __restrict__ att,
                         const __bf16* __restrict__ wot,
                         float* __restrict__ out) {
  int wave = threadIdx.x >> 5;
  int mt = wave >> 2, ntw = wave & 3;
  long row0 = (long)blockIdx.y * 64 + mt * 32;
  long col0 = (long)blockIdx.x * 256 + ntw * 64;

  v8f acc[2][4] = {};
  for (int k0 = 0; k0 < NHC; k0 += 32) {
    v16bf a0 = load_a_frag(att, row0, k0, NHC);
    v16bf a1 = load_a_frag(att, row0 + 16, k0, NHC);
    __builtin_prefetch(wot + col0 * NHC + k0 + 256, 0, 1);
#pragma unroll
    for (int j = 0; j < 4; ++j) {
      v16bf bfr = load_b_frag(wot, col0 + j * 16, k0, NHC);
      acc[0][j] = wmma_bf16(a0, bfr, acc[0][j]);
      acc[1][j] = wmma_bf16(a1, bfr, acc[1][j]);
    }
  }

  int lane = threadIdx.x & 31;
  int half = lane >> 4;
  int cidx = lane & 15;
#pragma unroll
  for (int i = 0; i < 2; ++i)
#pragma unroll
    for (int j = 0; j < 4; ++j)
#pragma unroll
      for (int r = 0; r < 8; ++r)
        out[(row0 + i * 16 + r + half * 8) * DIM + col0 + j * 16 + cidx] = acc[i][j][r];
}

// ---------------- host launch ----------------
extern "C" void kernel_launch(void* const* d_in, const int* in_sizes, int n_in,
                              void* d_out, int out_size, void* d_ws, size_t ws_size,
                              hipStream_t stream) {
  (void)in_sizes; (void)n_in; (void)out_size; (void)ws_size;
  const float* x  = (const float*)d_in[0];
  const float* Wq = (const float*)d_in[1];
  const float* Wk = (const float*)d_in[2];
  const float* Wv = (const float*)d_in[3];
  const float* Wo = (const float*)d_in[4];
  float* out = (float*)d_out;

  char* w = (char*)d_ws;
  __bf16* xb  = (__bf16*)w; w += (size_t)MTOT * DIM * 2;
  __bf16* wqt = (__bf16*)w; w += (size_t)NHC * DIM * 2;
  __bf16* wkt = (__bf16*)w; w += (size_t)NHC * DIM * 2;
  __bf16* wvt = (__bf16*)w; w += (size_t)NHC * DIM * 2;
  __bf16* wot = (__bf16*)w; w += (size_t)NHC * DIM * 2;
  __bf16* qb  = (__bf16*)w; w += (size_t)BATCH * NH * SEQ * HD * 2;
  __bf16* kb  = (__bf16*)w; w += (size_t)BATCH * NH * SEQ * HD * 2;
  __bf16* vT  = (__bf16*)w; w += (size_t)BATCH * NH * HD * SEQ * 2;
  __bf16* att = (__bf16*)w; w += (size_t)MTOT * NHC * 2;

  int nx = MTOT * DIM;
  cvt_f32_to_bf16<<<(nx + 255) / 256, 256, 0, stream>>>(x, xb, nx);

  dim3 tb(32, 8), tg(DIM / 32, DIM / 32);
  transpose_cvt<<<tg, tb, 0, stream>>>(Wq, wqt);
  transpose_cvt<<<tg, tb, 0, stream>>>(Wk, wkt);
  transpose_cvt<<<tg, tb, 0, stream>>>(Wv, wvt);
  transpose_cvt<<<tg, tb, 0, stream>>>(Wo, wot);

  qkv_proj<<<dim3(NHC / 256, MTOT / 64, 3), 256, 0, stream>>>(xb, wqt, wkt, wvt, qb, kb, vT);
  attention<<<dim3(SEQ / 128, BATCH * NH), 256, 0, stream>>>(qb, kb, vT, att);
  out_proj<<<dim3(DIM / 256, MTOT / 64), 256, 0, stream>>>(att, wot, out);
}
